// MOE_ViT_37933151158766
// MI455X (gfx1250) — compile-verified, matrix operand order checked by reading, not run on hardware
//
#include <hip/hip_runtime.h>
#include <hip/hip_bf16.h>

// ---------------------------------------------------------------------------
// MoE-ViT forward for MI455X (gfx1250): bf16 WMMA GEMMs, fp32 accumulate,
// bf16 activations, A-operand tiles streamed by the Tensor Data Mover into
// double-buffered LDS (one barrier per K-step, DMA overlapped with WMMA).
// E=4, B=16, S=224, P=16, D=384, H=6, HD=64, L=12, FF=1536, NC=10
// T=197 tokens -> 208 compute rows; activation buffers padded to 256
// rows/image so unguarded 128-row DMA tile reads stay in-bounds.
//
// LDS tiles are bf16 pair-packed u32, A and B both "row-major over K-pairs":
//   As32[m][kp] holds A[m, 2kp..2kp+1]; row stride 20 u32 (80 B) — exactly
//   what TDM pad_interval=16dw / pad_amount=4dw produces.
// ---------------------------------------------------------------------------

typedef __attribute__((ext_vector_type(16))) __bf16 v16bf;
typedef __attribute__((ext_vector_type(8)))  float  v8f;
typedef __attribute__((ext_vector_type(4)))  unsigned int v4u;
typedef __attribute__((ext_vector_type(8)))  int v8i32;
typedef __attribute__((ext_vector_type(4)))  int v4i32;

#define FLAG_GELU    1
#define FLAG_WBF16   2
#define FLAG_OUTBF16 4
#define FLAG_BIAS    8
#define FLAG_RES     16

union Frag { uint32_t u[8]; v16bf v; };

static __device__ __forceinline__ float gelu_exact(float x) {
    return 0.5f * x * (1.0f + erff(x * 0.7071067811865475f));
}

static __device__ __forceinline__ uint32_t pack_bf16(float a, float b) {
    union { __bf16 h[2]; uint32_t u; } p;
    p.h[0] = (__bf16)a; p.h[1] = (__bf16)b;
    return p.u;
}

// load a 16x32 bf16 fragment (A-style layout) from a pair-packed LDS row
static __device__ __forceinline__ v16bf load_frag(const uint32_t* row, int hi) {
    Frag f;
    *(uint4*)&f.u[0] = *(const uint4*)(row + hi * 4);
    *(uint4*)&f.u[4] = *(const uint4*)(row + 8 + hi * 4);
    return f.v;
}

// ---------------------------------------------------------------------------
// TDM: DMA a rows x 32-col bf16 tile from global [row][rowlen] into LDS,
// inserting 16B padding after every 64B row => 80B LDS row stride.
// D# group0/group1 built per CDNA5 ISA 8.3/8.4; 2D tensor: groups 2,3 unused.
// ---------------------------------------------------------------------------
static __device__ __forceinline__ void tdm_load_tile_2d(
    uint32_t lds_off, const __bf16* src, int rowlen, int rows)
{
    unsigned long long ga = (unsigned long long)(const void*)src;
    v4u g0;
    g0[0] = 1u;                                          // count=1 (valid user D#)
    g0[1] = lds_off;                                     // lds_addr (bytes)
    g0[2] = (unsigned)ga;                                // global_addr lo
    g0[3] = (unsigned)((ga >> 32) & 0x1FFFFFFu) | (2u << 30);  // addr hi | type=2
    v8i32 g1;
    g1[0] = (int)((1u << 16)      // data_size = 1 (2 bytes)
                | (1u << 20)      // pad_enable
                | (3u << 22)      // pad_interval: 16 DWORDs (64B) before pad
                | (3u << 25));    // pad_amount: 4 DWORDs (16B)
    g1[1] = (int)(((unsigned)rowlen & 0xFFFFu) << 16);   // tensor_dim0 lo16
    g1[2] = (int)(((unsigned)rowlen >> 16) & 0xFFFFu);   // tensor_dim0 hi16 | dim1 lo=0
    g1[3] = (int)(16u | (32u << 16));                    // tensor_dim1=1M | tile_dim0=32
    g1[4] = rows;                                        // tile_dim1 (tile_dim2=0)
    g1[5] = rowlen;                                      // tensor_dim0_stride lo32
    g1[6] = 0;
    g1[7] = 0;
    v4i32 z4 = (v4i32)0;
    v8i32 z8 = (v8i32)0;
    __builtin_amdgcn_tensor_load_to_lds(g0, g1, z4, z4, z8, 0);
}

// ---------------------------------------------------------------------------
// Gating: logits[b,e] = dot(x[b,:150528], gate_w[:,e]) + gate_b[e]
// ---------------------------------------------------------------------------
__global__ __launch_bounds__(256) void gate_kernel(
    const float* __restrict__ x, const float* __restrict__ gw,
    const float* __restrict__ gb, float* __restrict__ logits)
{
    const int b = blockIdx.x >> 2;
    const int e = blockIdx.x & 3;
    const float* xr = x + (long long)b * 150528;
    float s = 0.f;
    for (int i = threadIdx.x; i < 150528; i += 256)
        s += xr[i] * gw[(long long)i * 4 + e];
    __shared__ float red[256];
    red[threadIdx.x] = s;
    __syncthreads();
    for (int st = 128; st > 0; st >>= 1) {
        if (threadIdx.x < st) red[threadIdx.x] += red[threadIdx.x + st];
        __syncthreads();
    }
    if (threadIdx.x == 0) logits[blockIdx.x] = red[0] + gb[e];
}

__global__ void argmax_kernel(const float* __restrict__ logits, int* __restrict__ idx)
{
    int b = threadIdx.x;
    if (b >= 16) return;
    const float* l = logits + b * 4;
    int best = 0; float bv = l[0];
    for (int e = 1; e < 4; ++e) { if (l[e] > bv) { bv = l[e]; best = e; } }
    idx[b] = best;
}

// ---------------------------------------------------------------------------
// Generic WMMA GEMM: C[M,N] = epi(A_bf16[M,K] @ W[K,N] (+bias) (+Res))
// Block tile 128x64, 8 waves, each wave = 32(M) x 32(N) via 4 wmma accs.
// A staged by TDM into double-buffered LDS (pipelined one K-step ahead);
// W staged by VALU pack (fp32 or bf16 per FLAGS) into double-buffered LDS.
// One barrier per K-step. FLAGS is a template arg -> branch-free hot loop.
// A base = A + z*aBatch + (n0/64)*aHead    (aHead used for attn A*V)
// W base = W + z*wBatch + eidx[z]*wExpert
// ---------------------------------------------------------------------------
template<int FLAGS>
__global__ __launch_bounds__(256) void gemm_wmma_kernel(
    const __bf16* __restrict__ A, int lda, long long aBatch, long long aHead,
    const void* __restrict__ W, int ldw, long long wBatch, long long wExpert,
    const int* __restrict__ eidx,
    const float* __restrict__ bias, long long biasExpert,
    const float* __restrict__ Res, long long resBatch,
    void* __restrict__ C, int ldc, long long cBatch,
    int M, int N, int K)
{
    __shared__ uint32_t As32[2][128][20];  // double-buffered, TDM-filled
    __shared__ uint32_t Bs32[2][64][20];

    const int z  = blockIdx.z;
    const int m0 = blockIdx.y * 128;
    const int n0 = blockIdx.x * 64;
    const int tid = threadIdx.x;
    const int e = eidx ? eidx[z] : 0;

    const __bf16* Ab = A + (long long)z * aBatch + (long long)(n0 >> 6) * aHead;
    const long long wBase = (long long)z * wBatch + (long long)e * wExpert;

    const int lane = tid & 31;
    const int wid  = tid >> 5;
    const int wm   = wid & 3;     // 32-row subtile
    const int wn   = wid >> 2;    // 32-col subtile
    const int ml   = lane & 15;
    const int hi   = lane >> 4;

    v8f acc[2][2] = {};

    const int ksteps = (K + 31) >> 5;
    // prologue: DMA first A tile
    if (wid == 0) {
        tdm_load_tile_2d((uint32_t)(size_t)&As32[0][0][0],
                         Ab + (long long)m0 * lda, lda, 128);
    }
    for (int kt = 0; kt < ksteps; ++kt) {
        const int buf = kt & 1;
        const int kbase = kt << 5;
        // W tile: 64 cols x 16 k-pairs, coalesced across n per k row
        #pragma unroll
        for (int i = 0; i < 4; ++i) {
            int el = i * 256 + tid;
            int nn = el & 63, kp = el >> 6;
            long long boff = (long long)(kbase + 2 * kp) * ldw + (n0 + nn);
            uint32_t pv;
            if constexpr (FLAGS & FLAG_WBF16) {
                const __bf16* Wp = (const __bf16*)W + wBase;
                union { __bf16 h[2]; uint32_t u; } p;
                p.h[0] = Wp[boff]; p.h[1] = Wp[boff + ldw];
                pv = p.u;
                __builtin_prefetch(Wp + boff + 32LL * ldw, 0, 1);
            } else {
                const float* Wp = (const float*)W + wBase;
                pv = pack_bf16(Wp[boff], Wp[boff + ldw]);
                __builtin_prefetch(Wp + boff + 32LL * ldw, 0, 1);
            }
            Bs32[buf][nn][kp] = pv;
        }
        if (wid == 0) __builtin_amdgcn_s_wait_tensorcnt(0);
        __syncthreads();
        // pipeline: DMA next A tile into the other buffer (its last readers
        // finished before the barrier we just passed)
        if (wid == 0 && kt + 1 < ksteps) {
            tdm_load_tile_2d((uint32_t)(size_t)&As32[buf ^ 1][0][0],
                             Ab + (long long)m0 * lda + (kbase + 32), lda, 128);
        }

        v16bf af[2], bfr[2];
        af[0]  = load_frag(&As32[buf][wm * 32 + ml][0], hi);
        af[1]  = load_frag(&As32[buf][wm * 32 + 16 + ml][0], hi);
        bfr[0] = load_frag(&Bs32[buf][wn * 32 + ml][0], hi);
        bfr[1] = load_frag(&Bs32[buf][wn * 32 + 16 + ml][0], hi);
        #pragma unroll
        for (int s2 = 0; s2 < 2; ++s2)
            #pragma unroll
            for (int h2 = 0; h2 < 2; ++h2)
                acc[s2][h2] = __builtin_amdgcn_wmma_f32_16x16x32_bf16(
                    false, af[s2], false, bfr[h2], (short)0, acc[s2][h2], false, false);
    }

    // epilogue: C 16x16 layout — VGPR v holds row (v + 8*hi), col = lane&15
    const float* bb   = bias + (long long)e * biasExpert;
    const float* Resb = Res + (long long)z * resBatch;
    #pragma unroll
    for (int s2 = 0; s2 < 2; ++s2) {
        #pragma unroll
        for (int v = 0; v < 8; ++v) {
            int row = m0 + wm * 32 + s2 * 16 + v + hi * 8;
            if (row >= M) continue;
            #pragma unroll
            for (int h2 = 0; h2 < 2; ++h2) {
                int col = n0 + wn * 32 + h2 * 16 + ml;
                if (col >= N) continue;
                float xv = acc[s2][h2][v];
                if constexpr (FLAGS & FLAG_BIAS) xv += bb[col];
                if constexpr (FLAGS & FLAG_GELU) xv = gelu_exact(xv);
                if constexpr (FLAGS & FLAG_RES)  xv += Resb[(long long)row * ldc + col];
                long long coff = (long long)z * cBatch + (long long)row * ldc + col;
                if constexpr (FLAGS & FLAG_OUTBF16) ((__bf16*)C)[coff] = (__bf16)xv;
                else                                ((float*)C)[coff] = xv;
            }
        }
    }
}

// ---------------------------------------------------------------------------
// Attention scores: S[b,h,q,k] = scale * sum_d Q[b,q,h,d] * K[b,k,h,d]
// All four tiles (Q/K x 2 d-halves) DMA'd up front; one barrier total.
// ---------------------------------------------------------------------------
__global__ __launch_bounds__(256) void attn_scores_kernel(
    const __bf16* __restrict__ qkv, __bf16* __restrict__ scores, float scale)
{
    __shared__ uint32_t Qs32[2][64][20];  // [d-half][query][d-pair]
    __shared__ uint32_t Ks32[2][64][20];  // [d-half][key][d-pair]

    const int z = blockIdx.z;           // b*6 + h
    const int b = z / 6, h = z % 6;
    const __bf16* Q  = qkv + (long long)b * (256LL * 1152) + h * 64;
    const __bf16* Kp = Q + 384;
    __bf16* Sb = scores + (long long)z * (256LL * 224);

    const int m0 = blockIdx.y * 64;     // queries
    const int n0 = blockIdx.x * 64;     // keys
    const int tid = threadIdx.x;
    const int lane = tid & 31;
    const int wid  = tid >> 5;
    const int wm   = wid & 3;
    const int wn   = wid >> 2;
    const int ml   = lane & 15;
    const int hi   = lane >> 4;

    v8f acc0 = {}; v8f acc1 = {};

    if (wid == 0) {
        tdm_load_tile_2d((uint32_t)(size_t)&Qs32[0][0][0], Q  + (long long)m0 * 1152,      1152, 64);
        tdm_load_tile_2d((uint32_t)(size_t)&Ks32[0][0][0], Kp + (long long)n0 * 1152,      1152, 64);
        tdm_load_tile_2d((uint32_t)(size_t)&Qs32[1][0][0], Q  + (long long)m0 * 1152 + 32, 1152, 64);
        tdm_load_tile_2d((uint32_t)(size_t)&Ks32[1][0][0], Kp + (long long)n0 * 1152 + 32, 1152, 64);
        __builtin_amdgcn_s_wait_tensorcnt(0);
    }
    __syncthreads();

    #pragma unroll
    for (int kt = 0; kt < 2; ++kt) {    // HD = 64 -> 2 K-steps of 32
        v16bf af  = load_frag(&Qs32[kt][wm * 16 + ml][0], hi);
        v16bf bf0 = load_frag(&Ks32[kt][wn * 32 + ml][0], hi);
        v16bf bf1 = load_frag(&Ks32[kt][wn * 32 + 16 + ml][0], hi);
        acc0 = __builtin_amdgcn_wmma_f32_16x16x32_bf16(false, af, false, bf0,
                                                       (short)0, acc0, false, false);
        acc1 = __builtin_amdgcn_wmma_f32_16x16x32_bf16(false, af, false, bf1,
                                                       (short)0, acc1, false, false);
    }

    #pragma unroll
    for (int v = 0; v < 8; ++v) {
        int row = m0 + wm * 16 + v + hi * 8;
        if (row >= 208) continue;
        int col0 = n0 + wn * 32 + ml;
        int col1 = col0 + 16;
        if (col0 < 208) Sb[(long long)row * 224 + col0] = (__bf16)(acc0[v] * scale);
        if (col1 < 208) Sb[(long long)row * 224 + col1] = (__bf16)(acc1[v] * scale);
    }
}

// ---------------------------------------------------------------------------
// Row softmax: read bf16, fp32 math, write bf16; keys >=197 zeroed through
// the full 224-col padded row. One wave32 per row; grid (208, B*H).
// ---------------------------------------------------------------------------
__global__ __launch_bounds__(32) void softmax_kernel(__bf16* __restrict__ scores)
{
    __bf16* row = scores + ((long long)blockIdx.y * 256 + blockIdx.x) * 224;
    const int lane = threadIdx.x;
    float vals[7];
    float mx = -3.4e38f;
    #pragma unroll
    for (int j = 0; j < 7; ++j) {
        int i = j * 32 + lane;
        if (i < 197) { vals[j] = (float)row[i]; mx = fmaxf(mx, vals[j]); }
        else vals[j] = 0.f;
    }
    for (int o = 16; o > 0; o >>= 1) mx = fmaxf(mx, __shfl_xor(mx, o, 32));
    float s = 0.f;
    #pragma unroll
    for (int j = 0; j < 7; ++j) {
        int i = j * 32 + lane;
        if (i < 197) { float ex = __expf(vals[j] - mx); vals[j] = ex; s += ex; }
    }
    for (int o = 16; o > 0; o >>= 1) s += __shfl_xor(s, o, 32);
    const float inv = 1.f / s;
    #pragma unroll
    for (int j = 0; j < 7; ++j) {
        int i = j * 32 + lane;                       // covers all 224 cols
        row[i] = (i < 197) ? (__bf16)(vals[j] * inv) : (__bf16)0.f;
    }
}

// ---------------------------------------------------------------------------
// LayerNorm over D=384 (fp32 in -> bf16 out), one row per block.
// ---------------------------------------------------------------------------
__global__ __launch_bounds__(128) void layernorm_kernel(
    const float* __restrict__ in, const float* __restrict__ w,
    const float* __restrict__ bia, long long wExpert,
    const int* __restrict__ eidx, __bf16* __restrict__ out)
{
    const int b = blockIdx.y, t = blockIdx.x;
    const int e = eidx[b];
    const float* row = in + ((long long)b * 208 + t) * 384;
    __bf16* orow     = out + ((long long)b * 256 + t) * 384;
    const float* wr = w   + (long long)e * wExpert;
    const float* br = bia + (long long)e * wExpert;

    const int tx = threadIdx.x;
    float v0 = row[tx], v1 = row[tx + 128], v2 = row[tx + 256];
    __shared__ float red[128];
    red[tx] = v0 + v1 + v2;
    __syncthreads();
    for (int st = 64; st > 0; st >>= 1) {
        if (tx < st) red[tx] += red[tx + st];
        __syncthreads();
    }
    const float mean = red[0] * (1.0f / 384.0f);
    __syncthreads();
    float d0 = v0 - mean, d1 = v1 - mean, d2 = v2 - mean;
    red[tx] = d0 * d0 + d1 * d1 + d2 * d2;
    __syncthreads();
    for (int st = 64; st > 0; st >>= 1) {
        if (tx < st) red[tx] += red[tx + st];
        __syncthreads();
    }
    const float inv = rsqrtf(red[0] * (1.0f / 384.0f) + 1e-6f);
    orow[tx]       = (__bf16)(d0 * inv * wr[tx]       + br[tx]);
    orow[tx + 128] = (__bf16)(d1 * inv * wr[tx + 128] + br[tx + 128]);
    orow[tx + 256] = (__bf16)(d2 * inv * wr[tx + 256] + br[tx + 256]);
}

// ---------------------------------------------------------------------------
// Patch embedding: tok[b, 1+n, :] = im2col(x)[n, :768] @ patch_w[e] + patch_b[e]
// (fp32 image input, manual im2col staging, fp32 token output).
// ---------------------------------------------------------------------------
__global__ __launch_bounds__(256) void patch_embed_kernel(
    const float* __restrict__ x, const float* __restrict__ pw,
    const float* __restrict__ pb, const int* __restrict__ eidx,
    float* __restrict__ tok)
{
    __shared__ uint32_t As32[64][20];
    __shared__ uint32_t Bs32[64][20];

    const int b = blockIdx.z;
    const int e = eidx[b];
    const float* xb = x  + (long long)b * (3LL * 224 * 224);
    const float* Wb = pw + (long long)e * (768LL * 384);
    const float* bb = pb + e * 384;
    float* Cb = tok + (long long)b * (208LL * 384);

    const int m0 = blockIdx.y * 64;  // patch index
    const int n0 = blockIdx.x * 64;
    const int tid = threadIdx.x;
    const int lane = tid & 31;
    const int wid  = tid >> 5;
    const int wm   = wid & 3;
    const int wn   = wid >> 2;
    const int ml   = lane & 15;
    const int hi   = lane >> 4;

    v8f acc0 = {}; v8f acc1 = {};

    for (int kt = 0; kt < 24; ++kt) {  // K = 768
        const int kbase = kt << 5;
        #pragma unroll
        for (int i = 0; i < 4; ++i) {  // im2col stage (pairs within patch row)
            int el = i * 256 + tid;
            int kp = el & 15, mr = el >> 4;
            int patch = m0 + mr;
            int k = kbase + 2 * kp;
            int cp = min(patch, 195);
            int c  = k >> 8;
            int py = (k >> 4) & 15;
            int px = k & 15;
            int gy = cp / 14, gx = cp % 14;
            const float2 f = *(const float2*)(xb +
                ((long long)c * 224 + gy * 16 + py) * 224 + gx * 16 + px);
            As32[mr][kp] = (patch < 196) ? pack_bf16(f.x, f.y) : 0u;
        }
        #pragma unroll
        for (int i = 0; i < 4; ++i) {  // weights, fully in-bounds
            int el = i * 256 + tid;
            int nn = el & 63, kp = el >> 6;
            long long boff = (long long)(kbase + 2 * kp) * 384 + n0 + nn;
            Bs32[nn][kp] = pack_bf16(Wb[boff], Wb[boff + 384]);
        }
        __syncthreads();

        v16bf af  = load_frag(&As32[wm * 16 + ml][0], hi);
        v16bf bf0 = load_frag(&Bs32[wn * 32 + ml][0], hi);
        v16bf bf1 = load_frag(&Bs32[wn * 32 + 16 + ml][0], hi);
        acc0 = __builtin_amdgcn_wmma_f32_16x16x32_bf16(false, af, false, bf0,
                                                       (short)0, acc0, false, false);
        acc1 = __builtin_amdgcn_wmma_f32_16x16x32_bf16(false, af, false, bf1,
                                                       (short)0, acc1, false, false);
        __syncthreads();
    }

    #pragma unroll
    for (int v = 0; v < 8; ++v) {
        int patch = m0 + wm * 16 + v + hi * 8;
        if (patch >= 196) continue;            // tok row = patch + 1
        int col0 = n0 + wn * 32 + ml;
        int col1 = col0 + 16;
        Cb[(long long)(patch + 1) * 384 + col0] = acc0[v] + bb[col0];
        Cb[(long long)(patch + 1) * 384 + col1] = acc1[v] + bb[col1];
    }
}

// cls row + positional embedding + zero the padding rows [197, 208)
__global__ __launch_bounds__(384) void posadd_kernel(
    const float* __restrict__ cls, const float* __restrict__ pos,
    const int* __restrict__ eidx, float* __restrict__ tok)
{
    const int b = blockIdx.y, t = blockIdx.x, d = threadIdx.x;
    const int e = eidx[b];
    float* p = tok + ((long long)b * 208 + t) * 384 + d;
    if (t == 0)        *p  = cls[e * 384 + d] + pos[(long long)e * 197 * 384 + d];
    else if (t < 197)  *p += pos[(long long)e * 197 * 384 + (long long)t * 384 + d];
    else               *p  = 0.f;
}

// head: out[b,c] = fin[b,0,:] @ head_w[e] + head_b[e]   (one wave per class)
__global__ void head_kernel(
    const __bf16* __restrict__ fin, const float* __restrict__ hw,
    const float* __restrict__ hb, const int* __restrict__ eidx,
    float* __restrict__ out)
{
    const int b = blockIdx.x;
    const int c = threadIdx.y;
    const int lane = threadIdx.x;
    const int e = eidx[b];
    const __bf16* row = fin + (long long)b * 256 * 384;   // token 0
    const float* w    = hw + (long long)e * 384 * 10;
    float s = 0.f;
    for (int d = lane; d < 384; d += 32) s += (float)row[d] * w[d * 10 + c];
    for (int o = 16; o > 0; o >>= 1) s += __shfl_xor(s, o, 32);
    if (lane == 0) out[b * 10 + c] = s + hb[e * 10 + c];
}

// ---------------------------------------------------------------------------
extern "C" void kernel_launch(void* const* d_in, const int* in_sizes, int n_in,
                              void* d_out, int out_size, void* d_ws, size_t ws_size,
                              hipStream_t stream) {
    const float* x       = (const float*)d_in[0];
    const float* gate_w  = (const float*)d_in[1];
    const float* gate_b  = (const float*)d_in[2];
    const float* patch_w = (const float*)d_in[3];
    const float* patch_b = (const float*)d_in[4];
    const float* cls     = (const float*)d_in[5];
    const float* pos     = (const float*)d_in[6];
    const float* ln1_w   = (const float*)d_in[7];
    const float* ln1_b   = (const float*)d_in[8];
    const float* qkv_w   = (const float*)d_in[9];
    const float* qkv_b   = (const float*)d_in[10];
    const float* proj_w  = (const float*)d_in[11];
    const float* proj_b  = (const float*)d_in[12];
    const float* ln2_w   = (const float*)d_in[13];
    const float* ln2_b   = (const float*)d_in[14];
    const float* fc1_w   = (const float*)d_in[15];
    const float* fc1_b   = (const float*)d_in[16];
    const float* fc2_w   = (const float*)d_in[17];
    const float* fc2_b   = (const float*)d_in[18];
    const float* norm_w  = (const float*)d_in[19];
    const float* norm_b  = (const float*)d_in[20];
    const float* head_w  = (const float*)d_in[21];
    const float* head_b  = (const float*)d_in[22];
    float* out = (float*)d_out;

    // workspace carve-up (bytes)
    size_t off = 0;
    auto carve = [&](size_t nbytes) {
        void* p = (char*)d_ws + off;
        off = (off + nbytes + 255) & ~(size_t)255;
        return p;
    };
    float*  logits = (float*)carve(64 * 4);
    int*    idx    = (int*)carve(16 * 4);
    float*  tok    = (float*)carve(16LL * 208 * 384 * 4);      // fp32 residual stream
    __bf16* lnb    = (__bf16*)carve(16LL * 256 * 384 * 2);     // bf16, 256-row padded
    __bf16* qkv    = (__bf16*)carve(16LL * 256 * 1152 * 2);
    __bf16* scores = (__bf16*)carve(96LL * 256 * 224 * 2);
    __bf16* att    = (__bf16*)carve(16LL * 256 * 384 * 2);
    __bf16* ffh    = (__bf16*)carve(16LL * 256 * 1536 * 2);
    (void)ws_size; (void)in_sizes; (void)n_in; (void)out_size;

    // 1) gate + argmax
    gate_kernel<<<64, 256, 0, stream>>>(x, gate_w, gate_b, logits);
    argmax_kernel<<<1, 32, 0, stream>>>(logits, idx);

    // 2) patch embed + cls/pos
    patch_embed_kernel<<<dim3(6, 4, 16), 256, 0, stream>>>(x, patch_w, patch_b, idx, tok);
    posadd_kernel<<<dim3(208, 16), 384, 0, stream>>>(cls, pos, idx, tok);

    // 3) transformer layers
    for (int l = 0; l < 12; ++l) {
        layernorm_kernel<<<dim3(208, 16), 128, 0, stream>>>(
            tok, ln1_w + l * 384, ln1_b + l * 384, 12LL * 384, idx, lnb);

        // qkv(bf16) = ln @ qkv_w[e,l] + qkv_b
        gemm_wmma_kernel<FLAG_BIAS | FLAG_OUTBF16><<<dim3(18, 2, 16), 256, 0, stream>>>(
            lnb, 384, 256LL * 384, 0,
            qkv_w + (long long)l * 384 * 1152, 1152, 0, 12LL * 384 * 1152, idx,
            qkv_b + l * 1152, 12LL * 1152,
            nullptr, 0,
            qkv, 1152, 256LL * 1152,
            208, 1152, 384);

        attn_scores_kernel<<<dim3(4, 4, 96), 256, 0, stream>>>(qkv, scores, 0.125f);
        softmax_kernel<<<dim3(208, 96), 32, 0, stream>>>(scores);

        // att(bf16) = softmax(scores) @ V   (per-head A via aHead stride)
        gemm_wmma_kernel<FLAG_WBF16 | FLAG_OUTBF16><<<dim3(6, 2, 16), 256, 0, stream>>>(
            scores, 224, 6LL * 256 * 224, 256LL * 224,
            qkv + 768, 1152, 256LL * 1152, 0, nullptr,
            nullptr, 0,
            nullptr, 0,
            att, 384, 256LL * 384,
            208, 384, 208);

        // tok(fp32) = tok + att @ proj_w[e,l] + proj_b
        gemm_wmma_kernel<FLAG_BIAS | FLAG_RES><<<dim3(6, 2, 16), 256, 0, stream>>>(
            att, 384, 256LL * 384, 0,
            proj_w + (long long)l * 384 * 384, 384, 0, 12LL * 384 * 384, idx,
            proj_b + l * 384, 12LL * 384,
            tok, 208LL * 384,
            tok, 384, 208LL * 384,
            208, 384, 384);

        layernorm_kernel<<<dim3(208, 16), 128, 0, stream>>>(
            tok, ln2_w + l * 384, ln2_b + l * 384, 12LL * 384, idx, lnb);

        // ffh(bf16) = gelu(ln @ fc1_w + fc1_b)
        gemm_wmma_kernel<FLAG_BIAS | FLAG_GELU | FLAG_OUTBF16><<<dim3(24, 2, 16), 256, 0, stream>>>(
            lnb, 384, 256LL * 384, 0,
            fc1_w + (long long)l * 384 * 1536, 1536, 0, 12LL * 384 * 1536, idx,
            fc1_b + l * 1536, 12LL * 1536,
            nullptr, 0,
            ffh, 1536, 256LL * 1536,
            208, 1536, 384);

        // tok(fp32) = tok + ffh @ fc2_w + fc2_b
        gemm_wmma_kernel<FLAG_BIAS | FLAG_RES><<<dim3(6, 2, 16), 256, 0, stream>>>(
            ffh, 1536, 256LL * 1536, 0,
            fc2_w + (long long)l * 1536 * 384, 384, 0, 12LL * 1536 * 384, idx,
            fc2_b + l * 384, 12LL * 384,
            tok, 208LL * 384,
            tok, 384, 208LL * 384,
            208, 384, 1536);
    }

    // 4) final LN + head
    layernorm_kernel<<<dim3(208, 16), 128, 0, stream>>>(
        tok, norm_w, norm_b, 384, idx, lnb);
    head_kernel<<<16, dim3(32, 10), 0, stream>>>(lnb, head_w, head_b, idx, out);
}
